// Gate_48223892799903
// MI455X (gfx1250) — compile-verified, matrix operand order checked by reading, not run on hardware
//
#include <hip/hip_runtime.h>
#include <hip/hip_bf16.h>
#include <stdint.h>

typedef __attribute__((ext_vector_type(16))) __bf16        v16bf;
typedef __attribute__((ext_vector_type(8)))  float         v8f;
typedef __attribute__((ext_vector_type(8)))  unsigned int  v8u;

#define D_DIM       2048
#define N_EXP       64
#define TOPK_K      8
#define TOK_PER_BLK 128
#define LOG_STRIDE  65   // 64 + 1 pad -> conflict-free column access

__device__ __forceinline__ unsigned pack2_bf16(float lo, float hi) {
    unsigned a = __builtin_bit_cast(unsigned, lo);
    unsigned b = __builtin_bit_cast(unsigned, hi);
    return (b & 0xFFFF0000u) | (a >> 16);
}

// ---------------------------------------------------------------------------
// Kernel 0: W (f32) -> bf16 in workspace; W is 512KB and L2-resident after this.
// ---------------------------------------------------------------------------
__global__ void gate_wcvt_kernel(const float* __restrict__ W,
                                 unsigned short* __restrict__ Wb, int n) {
    int i = blockIdx.x * blockDim.x + threadIdx.x;
    if (i < n) Wb[i] = (unsigned short)(__builtin_bit_cast(unsigned, W[i]) >> 16);
}

// ---------------------------------------------------------------------------
// Kernel 1: fused router GEMM (WMMA bf16) + softmax + biased top-8 + partials
//   block = 256 threads (8 waves), 128 tokens per block.
//   Each wave: 16 tokens x 64 experts via 4x v_wmma_f32_16x16x32_bf16 accums.
// ---------------------------------------------------------------------------
__global__ void __launch_bounds__(256)
gate_main_kernel(const float* __restrict__ x,
                 const unsigned short* __restrict__ Wb,
                 const float* __restrict__ bias,
                 float* __restrict__ w_out,
                 int*   __restrict__ i_out,
                 int*   __restrict__ cnt_part,
                 float* __restrict__ psum_part,
                 int T) {
    __shared__ float s_log[TOK_PER_BLK * LOG_STRIDE];
    __shared__ float s_bias[N_EXP];
    __shared__ int   s_cnt[N_EXP];

    const int tid  = threadIdx.x;
    const int wv   = tid >> 5;        // wave id 0..7
    const int lane = tid & 31;
    const int col  = lane & 15;       // M (A rows / tokens) or N (B cols / experts)
    const int half = lane >> 4;       // selects K-chunk group per 16-bit WMMA layout

    if (tid < N_EXP) { s_bias[tid] = bias[tid]; s_cnt[tid] = 0; }

    // ---------------- Phase A: logits via WMMA ----------------
    const int tok0 = blockIdx.x * TOK_PER_BLK + wv * 16;
    const float* xrow = x + (size_t)(tok0 + col) * D_DIM;

    v8f zero8 = {0.f, 0.f, 0.f, 0.f, 0.f, 0.f, 0.f, 0.f};
    v8f acc[4] = {zero8, zero8, zero8, zero8};

    for (int k0 = 0; k0 < D_DIM; k0 += 32) {
        const int ka = k0 + half * 8;   // this half-wave's K chunks: ka..ka+7, ka+16..ka+23

        // prefetch next x chunk (emits global_prefetch_b8)
        __builtin_prefetch(xrow + ka + 32, 0, 1);

        // A fragment: 16 bf16 per lane, ISA 16-bit A layout
        const float4* pa = (const float4*)(xrow + ka);
        float4 a0 = pa[0];
        float4 a1 = pa[1];
        const float4* pa2 = (const float4*)(xrow + ka + 16);
        float4 a2 = pa2[0];
        float4 a3 = pa2[1];
        v8u au;
        au[0] = pack2_bf16(a0.x, a0.y);
        au[1] = pack2_bf16(a0.z, a0.w);
        au[2] = pack2_bf16(a1.x, a1.y);
        au[3] = pack2_bf16(a1.z, a1.w);
        au[4] = pack2_bf16(a2.x, a2.y);
        au[5] = pack2_bf16(a2.z, a2.w);
        au[6] = pack2_bf16(a3.x, a3.y);
        au[7] = pack2_bf16(a3.z, a3.w);
        v16bf afrag = __builtin_bit_cast(v16bf, au);

        // 4 expert tiles of 16: B columns are W rows (contiguous K, already packed)
        #pragma unroll
        for (int et = 0; et < 4; ++et) {
            const unsigned short* wr = Wb + (size_t)(et * 16 + col) * D_DIM + ka;
            const uint4 b0 = *(const uint4*)(wr);        // K = ka..ka+7
            const uint4 b1 = *(const uint4*)(wr + 16);   // K = ka+16..ka+23
            v8u bu;
            bu[0] = b0.x; bu[1] = b0.y; bu[2] = b0.z; bu[3] = b0.w;
            bu[4] = b1.x; bu[5] = b1.y; bu[6] = b1.z; bu[7] = b1.w;
            v16bf bfrag = __builtin_bit_cast(v16bf, bu);

            acc[et] = __builtin_amdgcn_wmma_f32_16x16x32_bf16(
                /*neg_a=*/false, afrag, /*neg_b=*/false, bfrag,
                /*c_mod=*/(short)0, acc[et], /*reuse_a=*/false, /*reuse_b=*/false);
        }
    }

    // C/D layout: VGPR i -> lanes 0-15: M=i, N=lane; lanes 16-31: M=8+i, N=lane-16
    #pragma unroll
    for (int et = 0; et < 4; ++et) {
        #pragma unroll
        for (int i = 0; i < 8; ++i) {
            const int tl = wv * 16 + half * 8 + i;
            s_log[tl * LOG_STRIDE + et * 16 + col] = acc[et][i];
        }
    }
    __syncthreads();

    // ---------------- Phase B: softmax + biased top-8 ----------------
    if (tid < TOK_PER_BLK) {
        float p[N_EXP];
        #pragma unroll
        for (int j = 0; j < N_EXP; ++j) p[j] = s_log[tid * LOG_STRIDE + j];

        float mx = -INFINITY;
        #pragma unroll
        for (int j = 0; j < N_EXP; ++j) mx = fmaxf(mx, p[j]);

        float sum = 0.f;
        #pragma unroll
        for (int j = 0; j < N_EXP; ++j) { p[j] = __expf(p[j] - mx); sum += p[j]; }
        const float inv = 1.0f / sum;
        #pragma unroll
        for (int j = 0; j < N_EXP; ++j) p[j] *= inv;

        const int tg = blockIdx.x * TOK_PER_BLK + tid;
        uint64_t taken = 0;
        for (int k = 0; k < TOPK_K; ++k) {
            float best  = -INFINITY;
            float bestw = 0.f;
            int   bi    = 0;
            #pragma unroll
            for (int j = 0; j < N_EXP; ++j) {
                const float v  = p[j] + s_bias[j];
                const bool  ok = (((taken >> j) & 1ull) == 0) && (v > best);
                best  = ok ? v    : best;
                bestw = ok ? p[j] : bestw;   // avoid dynamic p[bi] (no scratch spill)
                bi    = ok ? j    : bi;
            }
            taken |= (1ull << bi);
            w_out[(size_t)tg * TOPK_K + k] = bestw;
            i_out[(size_t)tg * TOPK_K + k] = bi;
            atomicAdd(&s_cnt[bi], 1);        // integer add: order-independent, exact
        }

        // write probs back for deterministic fixed-order column reduction
        #pragma unroll
        for (int j = 0; j < N_EXP; ++j) s_log[tid * LOG_STRIDE + j] = p[j];
    }
    __syncthreads();

    if (tid < N_EXP) {
        float s = 0.f;
        for (int t = 0; t < TOK_PER_BLK; ++t) s += s_log[t * LOG_STRIDE + tid];
        psum_part[(size_t)blockIdx.x * N_EXP + tid] = s;
        cnt_part [(size_t)blockIdx.x * N_EXP + tid] = s_cnt[tid];
    }
}

// ---------------------------------------------------------------------------
// Kernel 2: deterministic reduction of partials -> f_i, expert_probs
// ---------------------------------------------------------------------------
__global__ void gate_reduce_kernel(const int* __restrict__ cnt_part,
                                   const float* __restrict__ psum_part,
                                   float* __restrict__ f_out,
                                   float* __restrict__ p_out,
                                   int nblk, int T) {
    const int e = threadIdx.x;
    if (e < N_EXP) {
        long  c = 0;
        float s = 0.f;
        for (int b = 0; b < nblk; ++b) {
            c += cnt_part [(size_t)b * N_EXP + e];
            s += psum_part[(size_t)b * N_EXP + e];
        }
        f_out[e] = (float)c * (float)N_EXP / ((float)TOPK_K * (float)T + 1e-6f);
        p_out[e] = s / (float)T;
    }
}

extern "C" void kernel_launch(void* const* d_in, const int* in_sizes, int n_in,
                              void* d_out, int out_size, void* d_ws, size_t ws_size,
                              hipStream_t stream) {
    const float* x    = (const float*)d_in[0];   // [T, 2048]
    const float* W    = (const float*)d_in[1];   // [64, 2048]
    const float* bias = (const float*)d_in[2];   // [64]

    const int T    = in_sizes[0] / D_DIM;
    const int nblk = T / TOK_PER_BLK;

    float* out   = (float*)d_out;
    float* w_out = out;                                  // [T, 8] f32
    int*   i_out = (int*)(out + (size_t)T * TOPK_K);     // [T, 8] i32
    float* f_out = out + (size_t)T * TOPK_K * 2;         // [64]
    float* p_out = f_out + N_EXP;                        // [64]

    unsigned short* Wb = (unsigned short*)d_ws;          // 64*2048*2 = 256KB
    char*  ws2       = (char*)d_ws + (size_t)N_EXP * D_DIM * sizeof(unsigned short);
    int*   cnt_part  = (int*)ws2;                                  // [nblk, 64]
    float* psum_part = (float*)(ws2 + (size_t)nblk * N_EXP * 4);   // [nblk, 64]

    gate_wcvt_kernel<<<(N_EXP * D_DIM + 255) / 256, 256, 0, stream>>>(W, Wb, N_EXP * D_DIM);
    gate_main_kernel<<<nblk, 256, 0, stream>>>(x, Wb, bias, w_out, i_out,
                                               cnt_part, psum_part, T);
    gate_reduce_kernel<<<1, N_EXP, 0, stream>>>(cnt_part, psum_part, f_out, p_out, nblk, T);
}
